// KTBM_mat_pers_37323265802339
// MI455X (gfx1250) — compile-verified, matrix operand order checked by reading, not run on hardware
//
#include <hip/hip_runtime.h>
#include <hip/hip_bf16.h>
#include <math.h>

// ---------------------------------------------------------------------------
// KTBM (DKVMN + LSTM) fused persistent-scan implementation for gfx1250.
//   B=256, T=200.  Output: preds (B,T) float32, out[b*T+t].
//
//  * prep kernels: swizzle weights into WMMA-fragment-ordered f16; precompute
//    per-batch student constants.
//  * one persistent scan kernel: 16 WGs x 512 threads (16 waves), each WG owns
//    16 batch rows, iterates t=0..199.  Dense matmuls = v_wmma_f32_16x16x32_f16
//    (M=16 = the batch tile).  A operands staged in fragment-ordered f16 LDS
//    (written once by producers, consumed as raw ds_load_b128 pairs).  B
//    operands are plain loads of the swizzled f16 weights: the compiler pins
//    what fits in VGPRs and spills the rest to scratch once in the preamble
//    (steady-state reloads are L2-resident scratch_load_b128 - cheap).
//    Mv state lives in d_ws global scratch (8MB -> L2 resident on MI455X).
// ---------------------------------------------------------------------------

#define KB   256
#define KT   200

typedef __attribute__((ext_vector_type(16))) _Float16 v16h;
typedef __attribute__((ext_vector_type(8)))  float    v8f;

static __device__ __forceinline__ float sigf(float x) { return 1.0f / (1.0f + expf(-x)); }

// Fragment K pattern: lane-half hh, element e=2j+p -> k = ((j>>2)<<4)+(hh<<3)+((j&3)<<1)+p
// Inverse (writer side): given (m,k) produce the flat offset in a fragment-ordered
// f16 buffer laid out as [ktile][lane][16 halfs], lane = hh*16 + m.
static __device__ __forceinline__ int afrag_off(int m, int k) {
    int kk = k & 31;
    int hh = (kk >> 3) & 1;
    int j  = ((kk >> 4) << 2) | ((kk >> 1) & 3);
    int e  = (j << 1) | (kk & 1);
    return (((k >> 5) << 5) + (hh << 4) + m) * 16 + e;
}

// A fragment: contiguous 32B per lane from fragment-ordered f16 LDS.
static __device__ __forceinline__ v16h load_a16(const _Float16* base, int kt, int lane) {
    return *reinterpret_cast<const v16h*>(base + (((kt << 5) + lane) << 4));
}

// B fragment: contiguous 32B per lane from swizzled f16 weights in global.
static __device__ __forceinline__ v16h load_b16(const _Float16* __restrict__ wsw,
                                                int tile, int lane) {
    return *reinterpret_cast<const v16h*>(wsw + ((((size_t)tile) * 32 + lane) << 4));
}

static __device__ __forceinline__ v8f wmma_acc(v8f acc, v16h a, v16h b) {
    return __builtin_amdgcn_wmma_f32_16x16x32_f16(false, a, false, b, (short)0, acc,
                                                  false, false);
}

// ---------------------------------------------------------------------------
// Weight swizzler: W (KxN f32, row-major) -> fragment-ordered f16.
// ---------------------------------------------------------------------------
__global__ void ktbm_swizzle_b(const float* __restrict__ W, _Float16* __restrict__ out,
                               int K, int N) {
    int ntiles = N >> 4;
    int tile = blockIdx.x;
    int kt = tile / ntiles, nt = tile % ntiles;
    int lane = threadIdx.x;
    int n  = nt * 16 + (lane & 15);
    int hh = lane >> 4;
    _Float16* o = out + (((size_t)tile * 32 + lane) << 4);
#pragma unroll
    for (int j = 0; j < 8; ++j) {
        int k = kt * 32 + ((j >> 2) << 4) + (hh << 3) + ((j & 3) << 1);
        o[2 * j]     = (_Float16)W[(size_t)k * N + n];
        o[2 * j + 1] = (_Float16)W[(size_t)(k + 1) * N + n];
    }
}

// ---------------------------------------------------------------------------
// Per-batch student constants: erase_stu (B,128), add_stu (B,128), gate_const (B,512)
// ---------------------------------------------------------------------------
__global__ void ktbm_stu_const(const int* __restrict__ s_ids, const float* __restrict__ s_embed,
                               const float* __restrict__ We_s, const float* __restrict__ be_s,
                               const float* __restrict__ Wd_s, const float* __restrict__ Ws,
                               const float* __restrict__ bg,
                               float* __restrict__ es, float* __restrict__ ad,
                               float* __restrict__ gc) {
    int b = blockIdx.x;
    int n = threadIdx.x;            // 0..767
    const float* s = s_embed + (size_t)s_ids[b] * 64;
    if (n < 128) {
        float a = be_s[n];
        for (int c = 0; c < 64; ++c) a += s[c] * We_s[c * 128 + n];
        es[b * 128 + n] = a;
    } else if (n < 256) {
        int v = n - 128;
        float a = 0.f;
        for (int c = 0; c < 64; ++c) a += s[c] * Wd_s[c * 128 + v];
        ad[b * 128 + v] = a;
    } else {
        int j = n - 256;
        float a = bg[j];
        for (int c = 0; c < 64; ++c) a += s[c] * Ws[c * 512 + j];
        gc[b * 512 + j] = a;
    }
}

// ---------------------------------------------------------------------------
// Persistent fused scan kernel. grid=16, block=512 (16 waves). WG owns 16 b.
// ---------------------------------------------------------------------------
__global__ __launch_bounds__(512, 1)
void ktbm_scan(const int* __restrict__ q_dat, const int* __restrict__ a_dat,
               const int* __restrict__ d_dat, const int* __restrict__ s_ids,
               const float* __restrict__ q_embed, const float* __restrict__ a_embed,
               const float* __restrict__ q_corr,  const float* __restrict__ q_beh,
               const float* __restrict__ d_embed,
               const float* __restrict__ be_q, const float* __restrict__ bd_q,
               const float* __restrict__ bmap,
               const float* __restrict__ w_k2b, const float* __restrict__ b_k2b,
               const float* __restrict__ bo1,  const float* __restrict__ Wo2,
               const float* __restrict__ bo2,
               const float* __restrict__ M0,
               const float* __restrict__ esbuf, const float* __restrict__ adbuf,
               const float* __restrict__ gcbuf,
               const _Float16* __restrict__ swWeq, const _Float16* __restrict__ swWdq,
               const _Float16* __restrict__ swWebh, const _Float16* __restrict__ swWdbh,
               const _Float16* __restrict__ swWmap, const _Float16* __restrict__ swWg,
               const _Float16* __restrict__ swWo1,
               float* __restrict__ Mv, float* __restrict__ out) {
    // fragment-ordered f16 A-operand staging ([ktile][lane][16])
    __shared__ __align__(32) _Float16 fq_qe[4 * 512];   // q_emb   (K=128)
    __shared__ __align__(32) _Float16 fq_ae[2 * 512];   // a_emb   (K=64)
    __shared__ __align__(32) _Float16 fq_qb[2 * 512];   // q_beh   (K=64)
    __shared__ __align__(32) _Float16 fq_de[1 * 512];   // d_emb   (K=32)
    __shared__ __align__(32) _Float16 fq_xb[2 * 512];   // x_beh   (K=64)
    __shared__ __align__(32) _Float16 fq_k [4 * 512];   // k       (K=128)
    __shared__ __align__(32) _Float16 fq_rc[4 * 512];   // rc      (K=128)
    __shared__ __align__(32) _Float16 fq_h [4 * 512];   // h       (K=128)
    // f32 buffers
    __shared__ float sh_corr[16 * 64];  // softmaxed correlation weights
    __shared__ float sh_eq[16 * 128];   // erase pre-act -> e
    __shared__ float sh_aq[16 * 128];   // add pre-act -> ad
    __shared__ float sh_c[16 * 128];    // LSTM cell state
    __shared__ float sh_g[16 * 512];    // gates; later reused as o1 (16x128)

    const int tid  = threadIdx.x;
    const int lane = tid & 31;
    const int wv   = tid >> 5;          // 0..15
    const int lm   = lane & 15;
    const int hh   = lane >> 4;
    const int b0   = blockIdx.x * 16;

    // ---- init: h=c=0, Mv[b] = M0_stu[s_ids[b]] for our 16 rows -------------
    for (int i = tid; i < 16 * 128; i += 512) sh_c[i] = 0.f;
    for (int i = tid; i < 4 * 512; i += 512) fq_h[i] = (_Float16)0.f;
    for (int i = tid + 4 * 512 - 512; i < 4 * 512; i += 512) fq_h[i] = (_Float16)0.f;
    for (int i = tid; i < 2048; i += 512) fq_h[i] = (_Float16)0.f;
    for (int idx = tid; idx < 16 * 8192; idx += 512) {
        int m = idx >> 13, off = idx & 8191;
        Mv[(size_t)(b0 + m) * 8192 + off] = M0[(size_t)s_ids[b0 + m] * 8192 + off];
    }
    const float bk2b = b_k2b[0];
    __syncthreads();

    for (int t = 0; t < KT; ++t) {
        // ---- gather + corr softmax: wave wv handles local row m=wv --------
        {
            int m = wv, b = b0 + m;
            int qid = q_dat[b * KT + t];
            int aid = a_dat[b * KT + t];
            int did = d_dat[b * KT + t];
            const float* qe = q_embed + (size_t)qid * 128;
            for (int i = lane; i < 128; i += 32) fq_qe[afrag_off(m, i)] = (_Float16)qe[i];
            for (int i = lane; i < 64; i += 32) {
                fq_ae[afrag_off(m, i)] = (_Float16)a_embed[aid * 64 + i];
                fq_qb[afrag_off(m, i)] = (_Float16)q_beh[(size_t)qid * 64 + i];
            }
            fq_de[afrag_off(m, lane)] = (_Float16)d_embed[did * 32 + lane];
            float c0 = q_corr[(size_t)qid * 64 + lane];
            float c1 = q_corr[(size_t)qid * 64 + 32 + lane];
            float mx = fmaxf(c0, c1);
            for (int off = 16; off; off >>= 1) mx = fmaxf(mx, __shfl_xor(mx, off, 32));
            float e0 = expf(c0 - mx), e1 = expf(c1 - mx);
            float sm = e0 + e1;
            for (int off = 16; off; off >>= 1) sm += __shfl_xor(sm, off, 32);
            float inv = 1.f / sm;
            sh_corr[m * 64 + lane]      = e0 * inv;
            sh_corr[m * 64 + 32 + lane] = e1 * inv;
        }
        __syncthreads();

        // ---- phase 1 (WMMA): eq = qa@We_q + h@We_bh + be_q ; aq likewise;
        //                       x_beh = tanh([qb|d]@Wmap + bmap) ------------
        {
            bool is_e = (wv < 8);
            int nt = wv & 7;
            int n = nt * 16 + lm;
            float bias = is_e ? be_q[n] : bd_q[n];
            v8f a0, a1;
#pragma unroll
            for (int r = 0; r < 8; ++r) { a0[r] = bias; a1[r] = 0.f; }
            const _Float16* wq = is_e ? swWeq : swWdq;    // 6 ktiles x 8 ntiles
            const _Float16* wh = is_e ? swWebh : swWdbh;  // 4 ktiles x 8 ntiles
            // dual accumulators: even k-steps -> a0, odd -> a1 (breaks RAW chain)
            a0 = wmma_acc(a0, load_a16(fq_qe, 0, lane), load_b16(wq, 0 * 8 + nt, lane));
            a1 = wmma_acc(a1, load_a16(fq_qe, 1, lane), load_b16(wq, 1 * 8 + nt, lane));
            a0 = wmma_acc(a0, load_a16(fq_qe, 2, lane), load_b16(wq, 2 * 8 + nt, lane));
            a1 = wmma_acc(a1, load_a16(fq_qe, 3, lane), load_b16(wq, 3 * 8 + nt, lane));
            a0 = wmma_acc(a0, load_a16(fq_ae, 0, lane), load_b16(wq, 4 * 8 + nt, lane));
            a1 = wmma_acc(a1, load_a16(fq_ae, 1, lane), load_b16(wq, 5 * 8 + nt, lane));
            a0 = wmma_acc(a0, load_a16(fq_h, 0, lane), load_b16(wh, 0 * 8 + nt, lane));
            a1 = wmma_acc(a1, load_a16(fq_h, 1, lane), load_b16(wh, 1 * 8 + nt, lane));
            a0 = wmma_acc(a0, load_a16(fq_h, 2, lane), load_b16(wh, 2 * 8 + nt, lane));
            a1 = wmma_acc(a1, load_a16(fq_h, 3, lane), load_b16(wh, 3 * 8 + nt, lane));
            float* dst = is_e ? sh_eq : sh_aq;
#pragma unroll
            for (int r = 0; r < 8; ++r) dst[(hh * 8 + r) * 128 + n] = a0[r] + a1[r];
        }
        if (wv < 4) {   // x_beh: N=64 -> 4 ntiles, K=96 -> 3 ksteps
            int nt = wv;
            int n = nt * 16 + lm;
            v8f acc;
#pragma unroll
            for (int r = 0; r < 8; ++r) acc[r] = bmap[n];
            acc = wmma_acc(acc, load_a16(fq_qb, 0, lane), load_b16(swWmap, 0 * 4 + nt, lane));
            acc = wmma_acc(acc, load_a16(fq_qb, 1, lane), load_b16(swWmap, 1 * 4 + nt, lane));
            acc = wmma_acc(acc, load_a16(fq_de, 0, lane), load_b16(swWmap, 2 * 4 + nt, lane));
#pragma unroll
            for (int r = 0; r < 8; ++r)
                fq_xb[afrag_off(hh * 8 + r, n)] = (_Float16)tanhf(acc[r]);
        }
        __syncthreads();

        // ---- phase 2: e = sigmoid(eq + erase_stu), ad = tanh(aq + add_stu) -
        for (int i = tid; i < 16 * 128; i += 512) {
            int m = i >> 7, v = i & 127, b = b0 + m;
            sh_eq[i] = sigf(sh_eq[i] + esbuf[b * 128 + v]);
            sh_aq[i] = tanhf(sh_aq[i] + adbuf[b * 128 + v]);
        }
        __syncthreads();

        // ---- phase 3: Mv update + rc + k (L2-resident global state) --------
        {
            int v = tid & 127, bq = tid >> 7;   // 512 threads -> 4 (b,v) groups
#pragma unroll
            for (int g = 0; g < 4; ++g) {
                int m = bq * 4 + g, b = b0 + m;
                float e  = sh_eq[m * 128 + v];
                float ad = sh_aq[m * 128 + v];
                float* Mb = Mv + (size_t)b * 8192 + v;
                float rcacc = 0.f, kacc = 0.f;
                for (int c = 0; c < 64; ++c) {
                    float w  = sh_corr[m * 64 + c];
                    float mo = Mb[c * 128];
                    float mn = mo * (1.f - w * e) + w * ad;
                    Mb[c * 128] = mn;
                    rcacc += w * mn;
                    kacc  += w_k2b[c] * mn;
                }
                int fo = afrag_off(m, v);
                fq_rc[fo] = (_Float16)rcacc;
                fq_k[fo]  = (_Float16)(kacc + bk2b);
            }
        }
        __syncthreads();

        // ---- phase 4 (WMMA): gates = [x|k|h] @ [Wx;Wk;Wh] + gate_const ----
        {
            int nt0 = wv * 2, nt1 = nt0 + 1;   // two N tiles, shared A fragment
            int n0 = nt0 * 16 + lm, n1 = n0 + 16;
            v8f g0, g1;
#pragma unroll
            for (int r = 0; r < 8; ++r) {
                size_t row = (size_t)(b0 + hh * 8 + r) * 512;
                g0[r] = gcbuf[row + n0];
                g1[r] = gcbuf[row + n1];
            }
#pragma unroll
            for (int kt = 0; kt < 2; ++kt) {
                v16h a = load_a16(fq_xb, kt, lane);
                g0 = wmma_acc(g0, a, load_b16(swWg, kt * 32 + nt0, lane));
                g1 = wmma_acc(g1, a, load_b16(swWg, kt * 32 + nt1, lane));
            }
#pragma unroll
            for (int kt = 0; kt < 4; ++kt) {
                v16h a = load_a16(fq_k, kt, lane);
                g0 = wmma_acc(g0, a, load_b16(swWg, (2 + kt) * 32 + nt0, lane));
                g1 = wmma_acc(g1, a, load_b16(swWg, (2 + kt) * 32 + nt1, lane));
            }
#pragma unroll
            for (int kt = 0; kt < 4; ++kt) {
                v16h a = load_a16(fq_h, kt, lane);
                g0 = wmma_acc(g0, a, load_b16(swWg, (6 + kt) * 32 + nt0, lane));
                g1 = wmma_acc(g1, a, load_b16(swWg, (6 + kt) * 32 + nt1, lane));
            }
#pragma unroll
            for (int r = 0; r < 8; ++r) {
                int row = (hh * 8 + r) * 512;
                sh_g[row + n0] = g0[r];
                sh_g[row + n1] = g1[r];
            }
        }
        __syncthreads();

        // ---- phase 5: LSTM cell; h written straight to f16 fragment buffer -
        for (int i = tid; i < 16 * 128; i += 512) {
            int m = i >> 7, v = i & 127;
            float ig = sh_g[m * 512 + v];
            float gg = sh_g[m * 512 + 128 + v];
            float fg = sh_g[m * 512 + 256 + v];
            float og = sh_g[m * 512 + 384 + v];
            float cc = sigf(fg) * sh_c[i] + sigf(ig) * tanhf(gg);
            sh_c[i] = cc;
            fq_h[afrag_off(m, v)] = (_Float16)(sigf(og) * tanhf(cc));
        }
        __syncthreads();

        // ---- phase 6 (WMMA): o1 = tanh([qe|rc|h] @ Wo1 + bo1) -------------
        if (wv < 8) {
            int nt = wv;
            int n = nt * 16 + lm;
            v8f a0, a1;
#pragma unroll
            for (int r = 0; r < 8; ++r) { a0[r] = bo1[n]; a1[r] = 0.f; }
            a0 = wmma_acc(a0, load_a16(fq_qe, 0, lane), load_b16(swWo1, 0 * 8 + nt, lane));
            a1 = wmma_acc(a1, load_a16(fq_qe, 1, lane), load_b16(swWo1, 1 * 8 + nt, lane));
            a0 = wmma_acc(a0, load_a16(fq_qe, 2, lane), load_b16(swWo1, 2 * 8 + nt, lane));
            a1 = wmma_acc(a1, load_a16(fq_qe, 3, lane), load_b16(swWo1, 3 * 8 + nt, lane));
            a0 = wmma_acc(a0, load_a16(fq_rc, 0, lane), load_b16(swWo1, 4 * 8 + nt, lane));
            a1 = wmma_acc(a1, load_a16(fq_rc, 1, lane), load_b16(swWo1, 5 * 8 + nt, lane));
            a0 = wmma_acc(a0, load_a16(fq_rc, 2, lane), load_b16(swWo1, 6 * 8 + nt, lane));
            a1 = wmma_acc(a1, load_a16(fq_rc, 3, lane), load_b16(swWo1, 7 * 8 + nt, lane));
            a0 = wmma_acc(a0, load_a16(fq_h, 0, lane), load_b16(swWo1, 8 * 8 + nt, lane));
            a1 = wmma_acc(a1, load_a16(fq_h, 1, lane), load_b16(swWo1, 9 * 8 + nt, lane));
            a0 = wmma_acc(a0, load_a16(fq_h, 2, lane), load_b16(swWo1, 10 * 8 + nt, lane));
            a1 = wmma_acc(a1, load_a16(fq_h, 3, lane), load_b16(swWo1, 11 * 8 + nt, lane));
#pragma unroll
            for (int r = 0; r < 8; ++r)
                sh_g[(hh * 8 + r) * 128 + n] = tanhf(a0[r] + a1[r]);
        }
        __syncthreads();

        // ---- phase 7: logit + sigmoid -------------------------------------
        {
            int m = wv;
            float s = 0.f;
#pragma unroll
            for (int i = 0; i < 4; ++i) {
                int v = lane + i * 32;
                s += sh_g[m * 128 + v] * Wo2[v];
            }
            for (int off = 16; off; off >>= 1) s += __shfl_xor(s, off, 32);
            if (lane == 0) out[(size_t)(b0 + m) * KT + t] = sigf(s + bo2[0]);
        }
        __syncthreads();
    }
}

// ---------------------------------------------------------------------------
// Launch. Input order = setup_inputs() dict order (params expanded in
// insertion order): q_data, a_data, d_data, s_ids, then 29 params.
// ---------------------------------------------------------------------------
extern "C" void kernel_launch(void* const* d_in, const int* in_sizes, int n_in,
                              void* d_out, int out_size, void* d_ws, size_t ws_size,
                              hipStream_t stream) {
    const int*   q_data  = (const int*)d_in[0];
    const int*   a_data  = (const int*)d_in[1];
    const int*   d_data  = (const int*)d_in[2];
    const int*   s_ids   = (const int*)d_in[3];
    const float* q_embed = (const float*)d_in[4];
    const float* a_embed = (const float*)d_in[5];
    const float* q_corr  = (const float*)d_in[6];
    const float* q_beh   = (const float*)d_in[7];
    const float* d_embed = (const float*)d_in[8];
    const float* s_embed = (const float*)d_in[9];
    const float* We_q    = (const float*)d_in[10];
    const float* be_q    = (const float*)d_in[11];
    const float* We_bh   = (const float*)d_in[12];
    const float* We_s    = (const float*)d_in[13];
    const float* be_s    = (const float*)d_in[14];
    const float* Wd_q    = (const float*)d_in[15];
    const float* bd_q    = (const float*)d_in[16];
    const float* Wd_bh   = (const float*)d_in[17];
    const float* Wd_s    = (const float*)d_in[18];
    const float* Wmap    = (const float*)d_in[19];
    const float* bmap    = (const float*)d_in[20];
    const float* w_k2b   = (const float*)d_in[21];
    const float* b_k2b   = (const float*)d_in[22];
    const float* Wx      = (const float*)d_in[23];
    const float* Wk      = (const float*)d_in[24];
    const float* Wh      = (const float*)d_in[25];
    const float* Ws      = (const float*)d_in[26];
    const float* bg      = (const float*)d_in[27];
    const float* Wo1     = (const float*)d_in[28];
    const float* bo1     = (const float*)d_in[29];
    const float* Wo2     = (const float*)d_in[30];
    const float* bo2     = (const float*)d_in[31];
    const float* M0      = (const float*)d_in[32];

    // workspace layout (floats then halves; every region 32B aligned)
    float* ws_f = (float*)d_ws;
    float* Mv = ws_f;                          // 256*64*128 = 2,097,152 f
    float* es = Mv + 2097152;                  // 256*128
    float* ad = es + 32768;                    // 256*128
    float* gc = ad + 32768;                    // 256*512
    _Float16* h16    = (_Float16*)(gc + 131072);
    _Float16* swWeq  = h16;                    // 6*8*512
    _Float16* swWdq  = swWeq  + 24576;         // 6*8*512
    _Float16* swWebh = swWdq  + 24576;         // 4*8*512
    _Float16* swWdbh = swWebh + 16384;         // 4*8*512
    _Float16* swWmap = swWdbh + 16384;         // 3*4*512
    _Float16* swWg   = swWmap + 6144;          // 10*32*512 (Wx kt0-1, Wk kt2-5, Wh kt6-9)
    _Float16* swWo1  = swWg   + 163840;        // 12*8*512

    ktbm_swizzle_b<<<48, 32, 0, stream>>>(We_q,  swWeq,  192, 128);
    ktbm_swizzle_b<<<48, 32, 0, stream>>>(Wd_q,  swWdq,  192, 128);
    ktbm_swizzle_b<<<32, 32, 0, stream>>>(We_bh, swWebh, 128, 128);
    ktbm_swizzle_b<<<32, 32, 0, stream>>>(Wd_bh, swWdbh, 128, 128);
    ktbm_swizzle_b<<<12, 32, 0, stream>>>(Wmap,  swWmap,  96,  64);
    ktbm_swizzle_b<<<64, 32, 0, stream>>>(Wx,    swWg,               64, 512);
    ktbm_swizzle_b<<<128, 32, 0, stream>>>(Wk,   swWg + 2 * 32 * 512, 128, 512);
    ktbm_swizzle_b<<<128, 32, 0, stream>>>(Wh,   swWg + 6 * 32 * 512, 128, 512);
    ktbm_swizzle_b<<<96, 32, 0, stream>>>(Wo1,   swWo1, 384, 128);

    ktbm_stu_const<<<KB, 768, 0, stream>>>(s_ids, s_embed, We_s, be_s, Wd_s, Ws, bg,
                                           es, ad, gc);

    ktbm_scan<<<KB / 16, 512, 0, stream>>>(q_data, a_data, d_data, s_ids,
                                           q_embed, a_embed, q_corr, q_beh, d_embed,
                                           be_q, bd_q, bmap, w_k2b, b_k2b,
                                           bo1, Wo2, bo2, M0,
                                           es, ad, gc,
                                           swWeq, swWdq, swWebh, swWdbh, swWmap,
                                           swWg, swWo1,
                                           Mv, (float*)d_out);
}
